// Encoder_46119358825154
// MI455X (gfx1250) — matrix operand order repaired and checked by reading, not compiled, not run on hardware
//
#include <hip/hip_runtime.h>
#include <hip/hip_bf16.h>
#include <cstdint>
#include <cstddef>

// ---------------------------------------------------------------------------
// Transformer encoder (B=16,S=512,D=768,H=12,FF=3072,L=6) for gfx1250.
// GEMM core: v_wmma_f32_16x16x32_f16, A-tiles fed by the Tensor Data Mover
// (tensor_load_to_lds + s_wait_tensorcnt), double-buffered LDS.
// ---------------------------------------------------------------------------

typedef __attribute__((ext_vector_type(16))) _Float16 v16h;
typedef __attribute__((ext_vector_type(8)))  float    v8f;
typedef __attribute__((ext_vector_type(4)))  unsigned int u32x4;
typedef __attribute__((ext_vector_type(8)))  int      i32x8;
typedef __attribute__((ext_vector_type(4)))  int      i32x4;

static constexpr int Dm  = 768;
static constexpr int Hh  = 12;
static constexpr int FFd = 3072;
static constexpr int Ll  = 6;
static constexpr int Bb  = 16;
static constexpr int Ss  = 512;
static constexpr int NT  = Bb * Ss;   // 8192 tokens

#define DEV __device__ __forceinline__

union FragH { v16h v; uint4 q[2]; _Float16 h[16]; };

// A-fragment (16x32 f16): lanes 0..15 = rows M, half h -> K = 16g + h,
// loaded from row-major LDS.
DEV v16h lds_frag_a(const _Float16* base, int lda) {
  const int lane = threadIdx.x & 31;
  const int m = lane & 15, g = lane >> 4;
  const _Float16* p = base + m * lda + 16 * g;
  FragH u;
  u.q[0] = *(const uint4*)(p);
  u.q[1] = *(const uint4*)(p + 8);
  return u.v;
}

// B-fragment (32x16 f16): lanes 0..15 = cols N, half h -> K = 16g + h,
// loaded from LDS stored transposed [n][K].
DEV v16h lds_frag_b(const _Float16* base, int ldb) {
  const int lane = threadIdx.x & 31;
  const int n = lane & 15, g = lane >> 4;
  const _Float16* p = base + n * ldb + 16 * g;
  FragH u;
  u.q[0] = *(const uint4*)(p);
  u.q[1] = *(const uint4*)(p + 8);
  return u.v;
}

DEV v8f wmma16(v16h a, v16h b, v8f c) {
  return __builtin_amdgcn_wmma_f32_16x16x32_f16(
      false, a, false, b, (short)0, c, false, false);
}

// ---------------------------------------------------------------------------
// TDM: DMA a 128x32 f16 tile (row stride = K elements) from global into LDS.
// D# group0: count=1 | lds_addr | 57b global addr | type=2.
// D# group1: data_size=2B, tensor_dim0=K, tensor_dim1=M, tile=32x128,
//            tensor_dim0_stride=K.  2D tile -> groups 2/3 zero (NULL).
// ---------------------------------------------------------------------------
DEV void tdm_load_tile_a(const _Float16* gptr, _Float16* lptr, int K, int M) {
  const unsigned int lds = (unsigned int)(uintptr_t)lptr;   // addr[31:0] = LDS offset
  const unsigned long long ga = (unsigned long long)(uintptr_t)gptr;
  u32x4 g0;
  g0.x = 1u;                                            // count=1, user D#
  g0.y = lds;                                           // lds_addr
  g0.z = (unsigned int)(ga & 0xffffffffull);            // global_addr[31:0]
  g0.w = (unsigned int)((ga >> 32) & 0x01ffffffull) | (2u << 30);  // [56:32]|type=2
  const unsigned int uK = (unsigned int)K, uM = (unsigned int)M;
  i32x8 g1;
  g1[0] = (int)(1u << 16);                              // wg_mask=0, data_size=2B
  g1[1] = (int)((uK & 0xffffu) << 16);                  // tensor_dim0 lo16 @bit48
  g1[2] = (int)((uK >> 16) | ((uM & 0xffffu) << 16));   // dim0 hi16 | dim1 lo16
  g1[3] = (int)((uM >> 16) | (32u << 16));              // dim1 hi16 | tile_dim0=32
  g1[4] = (int)(128u);                                  // tile_dim1=128, tile_dim2=0
  g1[5] = (int)uK;                                      // tensor_dim0_stride lo32
  g1[6] = 0;                                            // stride hi | dim1_stride lo
  g1[7] = 0;
  const i32x4 z4 = {0, 0, 0, 0};
#if defined(__clang_major__) && __clang_major__ >= 23
  const i32x8 z8 = {0, 0, 0, 0, 0, 0, 0, 0};
  __builtin_amdgcn_tensor_load_to_lds(g0, g1, z4, z4, z8, 0);
#else
  __builtin_amdgcn_tensor_load_to_lds(g0, g1, z4, z4, 0);
#endif
}

// ---------------------------------------------------------------------------
// GEMM: C[M,N] = A[M,K]*B[K,N] + bias (f16 in, f32 acc).
// Block tile 128x128x32, 8 waves (2x4), 64x32 per wave = 8 WMMA / K-step.
// A-tile via TDM (wave 0), B-tile via manual LDS transpose; double-buffered.
// ---------------------------------------------------------------------------
template<int RELU, int WRITE_F32>
__global__ __launch_bounds__(256) void gemm_wmma(
    const _Float16* __restrict__ A, const _Float16* __restrict__ Bw,
    const float* __restrict__ bias, void* __restrict__ Cout,
    int M, int N, int K)
{
  __shared__ __align__(16) _Float16 As[2][128 * 32];   // [m][k]
  __shared__ __align__(16) _Float16 Bs[2][128 * 32];   // [n][k] transposed

  const int tid  = threadIdx.x;
  const int wave = tid >> 5;
  const int wm = wave >> 2, wn = wave & 3;
  const int bm0 = blockIdx.y * 128, bn0 = blockIdx.x * 128;
  const int lane = tid & 31;
  const int rsel = lane >> 4, csel = lane & 15;

  v8f acc[4][2];
#pragma unroll
  for (int i = 0; i < 4; i++)
#pragma unroll
    for (int j = 0; j < 2; j++) acc[i][j] = (v8f){};

  const int bk = tid >> 3, nb = (tid & 7) * 16;   // B-tile loader coords

  // Stage tile(s=0) into buffer 0.
  if (tid < 32)
    tdm_load_tile_a(A + (size_t)bm0 * K, &As[0][0], K, M);
  {
    FragH u;
    const uint4* src = (const uint4*)(Bw + (size_t)bk * N + bn0 + nb);
    u.q[0] = src[0]; u.q[1] = src[1];
#pragma unroll
    for (int j = 0; j < 16; j++) Bs[0][(nb + j) * 32 + bk] = u.h[j];
  }

  const int nsteps = K >> 5;
  for (int s = 0; s < nsteps; s++) {
    const int cur = s & 1;
    const int k0n = (s + 1) << 5;
    if (s + 1 < nsteps) {
      // Prefetch next tiles into the other buffer.
      if (tid < 32)
        tdm_load_tile_a(A + (size_t)bm0 * K + k0n, &As[cur ^ 1][0], K, M);
      FragH u;
      const uint4* src = (const uint4*)(Bw + (size_t)(k0n + bk) * N + bn0 + nb);
      u.q[0] = src[0]; u.q[1] = src[1];
#pragma unroll
      for (int j = 0; j < 16; j++) Bs[cur ^ 1][(nb + j) * 32 + bk] = u.h[j];
      if (s + 2 < nsteps)   // warm L2 two tiles ahead
        __builtin_prefetch(Bw + (size_t)(k0n + 32 + bk) * N + bn0 + nb, 0, 1);
      __builtin_amdgcn_s_wait_tensorcnt(1);   // current TDM done, next in flight
    } else {
      __builtin_amdgcn_s_wait_tensorcnt(0);
    }
    __syncthreads();   // publish As[cur] (TDM) + Bs[cur] (ds stores)

    v16h af[4], bf[2];
#pragma unroll
    for (int i = 0; i < 4; i++)
      af[i] = lds_frag_a(&As[cur][(wm * 64 + i * 16) * 32], 32);
#pragma unroll
    for (int j = 0; j < 2; j++)
      bf[j] = lds_frag_b(&Bs[cur][(wn * 32 + j * 16) * 32], 32);
#pragma unroll
    for (int i = 0; i < 4; i++)
#pragma unroll
      for (int j = 0; j < 2; j++)
        acc[i][j] = wmma16(af[i], bf[j], acc[i][j]);
    __syncthreads();   // protect buffers before next prefetch
  }

  // Straight-line epilogue (templated: no data-dependent branches).
#pragma unroll
  for (int i = 0; i < 4; i++)
#pragma unroll
    for (int j = 0; j < 2; j++)
#pragma unroll
      for (int r = 0; r < 8; r++) {
        const int row = bm0 + wm * 64 + i * 16 + r + 8 * rsel;
        const int col = bn0 + wn * 32 + j * 16 + csel;
        float v = acc[i][j][r] + bias[col];
        if (RELU) v = fmaxf(v, 0.0f);
        const size_t idx = (size_t)row * N + col;
        if (WRITE_F32) ((float*)Cout)[idx] = v;
        else           ((_Float16*)Cout)[idx] = (_Float16)v;
      }
}

// ---------------------------------------------------------------------------
// Attention: one block per (b, h, 16-row query tile). 128 threads = 4 waves.
// ---------------------------------------------------------------------------
__global__ __launch_bounds__(128) void attn_kernel(
    const _Float16* __restrict__ Q, const _Float16* __restrict__ Kh,
    const _Float16* __restrict__ Vh, const int* __restrict__ tokens,
    _Float16* __restrict__ O)
{
  __shared__ __align__(16) _Float16 Qs[16 * 64];
  __shared__ __align__(16) float    Sc[16 * 512];
  __shared__ __align__(16) _Float16 Ph[16 * 512];
  __shared__ float red[16 * 8];

  const int tid = threadIdx.x;
  const int lane = tid & 31;
  const int wave = tid >> 5;
  const int g = lane >> 4, nlo = lane & 15;
  const int qt = blockIdx.x, hh = blockIdx.y, bb = blockIdx.z;
  const int q0 = qt * 16;

  {   // load Q tile [16 x 64] f16
    const int q = tid >> 3, c = (tid & 7) * 8;
    const uint4* src =
        (const uint4*)(Q + (size_t)(bb * Ss + q0 + q) * Dm + hh * 64 + c);
    *(uint4*)(Qs + q * 64 + c) = *src;
  }
  __syncthreads();

  // ---- Phase 1: logits = Q K^T / 8 + mask -----------------------------
  for (int kb = wave * 16; kb < Ss; kb += 64) {
    v8f acc = (v8f){};
#pragma unroll
    for (int ks = 0; ks < 2; ks++) {
      v16h a = lds_frag_a(Qs + ks * 32, 64);
      FragH u;
      const _Float16* kp = Kh + (size_t)(bb * Ss + kb + nlo) * Dm +
                           hh * 64 + ks * 32 + 16 * g;
      u.q[0] = *(const uint4*)(kp);
      u.q[1] = *(const uint4*)(kp + 8);
      acc = wmma16(a, u.v, acc);
    }
    const int col = kb + nlo;
    const float mk = (tokens[bb * Ss + col] == 0) ? -1e9f : 0.0f;
#pragma unroll
    for (int r = 0; r < 8; r++)
      Sc[(r + 8 * g) * 512 + col] = acc[r] * 0.125f + mk;
  }
  __syncthreads();

  // ---- Phase 2: softmax ------------------------------------------------
  {
    const int row = tid >> 3, seg = tid & 7;
    float* p = Sc + row * 512 + seg * 64;
    float mx = -1e30f;
    for (int j = 0; j < 64; j++) mx = fmaxf(mx, p[j]);
    red[row * 8 + seg] = mx;
    __syncthreads();
    float m = red[row * 8 + 0];
#pragma unroll
    for (int k = 1; k < 8; k++) m = fmaxf(m, red[row * 8 + k]);
    float s = 0.0f;
    for (int j = 0; j < 64; j++) { float e = __expf(p[j] - m); p[j] = e; s += e; }
    __syncthreads();
    red[row * 8 + seg] = s;
    __syncthreads();
    float tot = 0.0f;
#pragma unroll
    for (int k = 0; k < 8; k++) tot += red[row * 8 + k];
    const float inv = 1.0f / tot;
    for (int j = 0; j < 64; j++)
      Ph[row * 512 + seg * 64 + j] = (_Float16)(p[j] * inv);
  }
  __syncthreads();

  // ---- Phase 3: O = P * V ---------------------------------------------
  v8f acc = (v8f){};
  const int dcol = wave * 16 + nlo;
  for (int ks = 0; ks < 16; ks++) {
    v16h a = lds_frag_a(Ph + ks * 32, 512);
    FragH u;
#pragma unroll
    for (int j = 0; j < 16; j++) {
      const int kt = ks * 32 + 16 * g + j;
      u.h[j] = Vh[(size_t)(bb * Ss + kt) * Dm + hh * 64 + dcol];
    }
    acc = wmma16(a, u.v, acc);
  }
#pragma unroll
  for (int r = 0; r < 8; r++) {
    const int row = q0 + r + 8 * g;
    O[(size_t)(bb * Ss + row) * Dm + hh * 64 + dcol] = (_Float16)acc[r];
  }
}

// ---------------------------------------------------------------------------
// Residual add + LayerNorm -> f32 + f16 shadow.
// ---------------------------------------------------------------------------
__global__ __launch_bounds__(256) void add_ln_kernel(
    const float* __restrict__ X, const float* __restrict__ Y,
    const float* __restrict__ G, const float* __restrict__ Bt,
    float* __restrict__ OutF, _Float16* __restrict__ OutH)
{
  __shared__ float s1[256], s2[256];
  const int row = blockIdx.x, t = threadIdx.x;
  const float* xp = X + (size_t)row * Dm;
  const float* yp = Y + (size_t)row * Dm;
  float v[3]; float sum = 0.0f, sq = 0.0f;
#pragma unroll
  for (int j = 0; j < 3; j++) {
    const int d = t + j * 256;
    v[j] = xp[d] + yp[d];
    sum += v[j]; sq += v[j] * v[j];
  }
  s1[t] = sum; s2[t] = sq;
  __syncthreads();
  for (int off = 128; off > 0; off >>= 1) {
    if (t < off) { s1[t] += s1[t + off]; s2[t] += s2[t + off]; }
    __syncthreads();
  }
  const float mean = s1[0] * (1.0f / (float)Dm);
  const float var  = s2[0] * (1.0f / (float)Dm) - mean * mean;
  const float inv  = rsqrtf(var + 1e-6f);
#pragma unroll
  for (int j = 0; j < 3; j++) {
    const int d = t + j * 256;
    const float o = (v[j] - mean) * inv * G[d] + Bt[d];
    OutF[(size_t)row * Dm + d] = o;
    OutH[(size_t)row * Dm + d] = (_Float16)o;
  }
}

// ---------------------------------------------------------------------------
// Embedding + sinusoidal positional encoding.
// ---------------------------------------------------------------------------
__global__ __launch_bounds__(256) void embed_kernel(
    const int* __restrict__ tokens, const float* __restrict__ emb,
    float* __restrict__ X, _Float16* __restrict__ Xh)
{
  const int row = blockIdx.x, t = threadIdx.x;
  const int tok = tokens[row];
  const float fs = (float)(row & (Ss - 1));
#pragma unroll
  for (int j = 0; j < 3; j++) {
    const int d = t + j * 256;
    const float rate = __powf(10000.0f, -(float)(2 * (d >> 1)) / (float)Dm);
    const float ang = fs * rate;
    const float pe = (d & 1) ? __cosf(ang) : __sinf(ang);
    const float val = emb[(size_t)tok * Dm + d] * 27.712812921102035f + pe;
    X[(size_t)row * Dm + d] = val;
    Xh[(size_t)row * Dm + d] = (_Float16)val;
  }
}

__global__ void f32_to_f16(const float* __restrict__ in,
                           _Float16* __restrict__ out, size_t n) {
  size_t i = (size_t)blockIdx.x * blockDim.x + threadIdx.x;
  const size_t stride = (size_t)gridDim.x * blockDim.x;
  for (; i < n; i += stride) out[i] = (_Float16)in[i];
}

// ---------------------------------------------------------------------------
extern "C" void kernel_launch(void* const* d_in, const int* in_sizes, int n_in,
                              void* d_out, int out_size, void* d_ws, size_t ws_size,
                              hipStream_t stream) {
  (void)in_sizes; (void)n_in; (void)out_size; (void)ws_size;
  const int*   tokens = (const int*)d_in[0];
  const float* emb  = (const float*)d_in[1];
  const float* wq   = (const float*)d_in[2];
  const float* bq   = (const float*)d_in[3];
  const float* wk   = (const float*)d_in[4];
  const float* bk   = (const float*)d_in[5];
  const float* wv   = (const float*)d_in[6];
  const float* bv   = (const float*)d_in[7];
  const float* wo   = (const float*)d_in[8];
  const float* bo   = (const float*)d_in[9];
  const float* w1   = (const float*)d_in[10];
  const float* b1   = (const float*)d_in[11];
  const float* w2   = (const float*)d_in[12];
  const float* b2   = (const float*)d_in[13];
  const float* ln1g = (const float*)d_in[14];
  const float* ln1b = (const float*)d_in[15];
  const float* ln2g = (const float*)d_in[16];
  const float* ln2b = (const float*)d_in[17];

  // Workspace carve-up (~286 MB total).
  char* p = (char*)d_ws;
  auto alloc = [&](size_t bytes) -> void* {
    void* r = (void*)p; p += (bytes + 255) & ~(size_t)255; return r;
  };
  float*    x   = (float*)   alloc((size_t)NT * Dm * 4);
  float*    y   = (float*)   alloc((size_t)NT * Dm * 4);
  float*    x1  = (float*)   alloc((size_t)NT * Dm * 4);
  _Float16* xh  = (_Float16*)alloc((size_t)NT * Dm * 2);
  _Float16* x1h = (_Float16*)alloc((size_t)NT * Dm * 2);
  _Float16* qh  = (_Float16*)alloc((size_t)NT * Dm * 2);
  _Float16* kh  = (_Float16*)alloc((size_t)NT * Dm * 2);
  _Float16* vh  = (_Float16*)alloc((size_t)NT * Dm * 2);
  _Float16* oh  = (_Float16*)alloc((size_t)NT * Dm * 2);
  _Float16* ffh = (_Float16*)alloc((size_t)NT * FFd * 2);
  _Float16* wqh = (_Float16*)alloc((size_t)Ll * Dm * Dm * 2);
  _Float16* wkh = (_Float16*)alloc((size_t)Ll * Dm * Dm * 2);
  _Float16* wvh = (_Float16*)alloc((size_t)Ll * Dm * Dm * 2);
  _Float16* woh = (_Float16*)alloc((size_t)Ll * Dm * Dm * 2);
  _Float16* w1h = (_Float16*)alloc((size_t)Ll * Dm * FFd * 2);
  _Float16* w2h = (_Float16*)alloc((size_t)Ll * FFd * Dm * 2);

  auto conv = [&](const float* src, _Float16* dst, size_t n) {
    f32_to_f16<<<dim3(4096), dim3(256), 0, stream>>>(src, dst, n);
  };
  conv(wq, wqh, (size_t)Ll * Dm * Dm);
  conv(wk, wkh, (size_t)Ll * Dm * Dm);
  conv(wv, wvh, (size_t)Ll * Dm * Dm);
  conv(wo, woh, (size_t)Ll * Dm * Dm);
  conv(w1, w1h, (size_t)Ll * Dm * FFd);
  conv(w2, w2h, (size_t)Ll * FFd * Dm);

  embed_kernel<<<dim3(NT), dim3(256), 0, stream>>>(tokens, emb, x, xh);

  const dim3 gD(Dm / 128, NT / 128);     // (6, 64)
  const dim3 gF(FFd / 128, NT / 128);    // (24, 64)
  const dim3 gA(Ss / 16, Hh, Bb);        // (32, 12, 16)

  for (int l = 0; l < Ll; l++) {
    const _Float16* wq_l = wqh + (size_t)l * Dm * Dm;
    const _Float16* wk_l = wkh + (size_t)l * Dm * Dm;
    const _Float16* wv_l = wvh + (size_t)l * Dm * Dm;
    const _Float16* wo_l = woh + (size_t)l * Dm * Dm;
    const _Float16* w1_l = w1h + (size_t)l * Dm * FFd;
    const _Float16* w2_l = w2h + (size_t)l * FFd * Dm;

    gemm_wmma<0,0><<<gD, 256, 0, stream>>>(xh, wq_l, bq + l * Dm, qh,
                                           NT, Dm, Dm);
    gemm_wmma<0,0><<<gD, 256, 0, stream>>>(xh, wk_l, bk + l * Dm, kh,
                                           NT, Dm, Dm);
    gemm_wmma<0,0><<<gD, 256, 0, stream>>>(xh, wv_l, bv + l * Dm, vh,
                                           NT, Dm, Dm);
    attn_kernel<<<gA, 128, 0, stream>>>(qh, kh, vh, tokens, oh);
    gemm_wmma<0,1><<<gD, 256, 0, stream>>>(oh, wo_l, bo + l * Dm, y,
                                           NT, Dm, Dm);
    add_ln_kernel<<<NT, 256, 0, stream>>>(x, y, ln1g + l * Dm, ln1b + l * Dm,
                                          x1, x1h);
    gemm_wmma<1,0><<<gF, 256, 0, stream>>>(x1h, w1_l, b1 + l * FFd, ffh,
                                           NT, FFd, Dm);
    gemm_wmma<0,1><<<gD, 256, 0, stream>>>(ffh, w2_l, b2 + l * Dm, y,
                                           NT, Dm, FFd);
    float* outp = (l == Ll - 1) ? (float*)d_out : x;
    add_ln_kernel<<<NT, 256, 0, stream>>>(x1, y, ln2g + l * Dm, ln2b + l * Dm,
                                          outp, xh);
  }
}
